// Qwen3_5MLP_88630945120707
// MI455X (gfx1250) — matrix operand order repaired and checked by reading, not compiled, not run on hardware
//
#include <hip/hip_runtime.h>

// ---------------------------------------------------------------------------
// Qwen3.5 LoRA MLP for gfx1250 (MI455X): bf16 WMMA GEMMs, f32 accumulate,
// async-to-LDS activation staging + TDM weight staging, double-buffered LDS.
// T=8192 tokens, D=2048, I=6144, NA=8 adapters, R=16 (NR = NA*R = 128).
// ---------------------------------------------------------------------------

typedef __bf16 bf16_t;
typedef __attribute__((ext_vector_type(8)))  __bf16 v8bf;
typedef __attribute__((ext_vector_type(16))) __bf16 v16bf;
typedef __attribute__((ext_vector_type(8)))  float  v8f;
typedef __attribute__((ext_vector_type(4)))  unsigned int u32x4;
typedef __attribute__((ext_vector_type(8)))  int i32x8;
typedef __attribute__((ext_vector_type(4)))  int i32x4;

constexpr int T_TOK = 8192;
constexpr int DIM_D = 2048;
constexpr int DIM_I = 6144;
constexpr int TWO_I = 12288;
constexpr int NR    = 128;   // NA * R
constexpr int LS    = 40;    // LDS row stride (bf16 elems) = 80B; 16 DW data + 4 DW pad

#if defined(__has_builtin)
#if __has_builtin(__builtin_amdgcn_tensor_load_to_lds)
#define HAVE_TDM 1
#endif
#endif

__device__ inline v8f wmma_bf16(v16bf a, v16bf b, v8f c) {
  return __builtin_amdgcn_wmma_f32_16x16x32_bf16(false, a, false, b, (short)0, c, false, false);
}

// A fragment (16x32 bf16): lanes 0-15 -> M=lane, K 0-7 & 16-23; lanes 16-31 -> K 8-15 & 24-31.
__device__ inline v16bf load_a_frag(const bf16_t* a, int row0, int lane) {
  int r  = row0 + (lane & 15);
  int kb = (lane >> 4) * 8;
  const bf16_t* p = a + r * LS + kb;
  v8bf lo = *(const v8bf*)p;
  v8bf hi = *(const v8bf*)(p + 16);
  return __builtin_shufflevector(lo, hi, 0,1,2,3,4,5,6,7,8,9,10,11,12,13,14,15);
}

// B fragment (32x16 bf16), tile stored N-major: lanes 0-15 -> N=lane, K 0-7 & 16-23; 16-31 -> K 8-15 & 24-31.
__device__ inline v16bf load_b_frag(const bf16_t* b, int col0, int lane) {
  int c  = col0 + (lane & 15);
  int kb = (lane >> 4) * 8;
  const bf16_t* p = b + c * LS + kb;
  v8bf lo = *(const v8bf*)p;
  v8bf hi = *(const v8bf*)(p + 16);
  return __builtin_shufflevector(lo, hi, 0,1,2,3,4,5,6,7,8,9,10,11,12,13,14,15);
}

// Async (ASYNCcnt-tracked) stage of rows x 32 bf16 panel into LDS (stride LS).
__device__ inline void stage_async(bf16_t* lds, const bf16_t* g, long ld, int rows, int tid) {
  int total = rows * 4;  // 16B chunks
  for (int c = tid; c < total; c += 256) {
    int r = c >> 2, q = c & 3;
    unsigned int laddr = (unsigned int)(size_t)(lds + r * LS + q * 8);
    const bf16_t* ga = g + (long)r * ld + q * 8;
    asm volatile("global_load_async_to_lds_b128 %0, %1, off"
                 :: "v"(laddr), "v"(ga) : "memory");
  }
}

__device__ inline void wait_async() {
  asm volatile("s_wait_asynccnt 0x0" ::: "memory");
}

// Synchronous fallback staging
__device__ inline void stage_sync(bf16_t* lds, const bf16_t* g, long ld, int rows, int tid) {
  int total = rows * 4;
  for (int c = tid; c < total; c += 256) {
    int r = c >> 2, q = c & 3;
    *(v8bf*)(lds + r * LS + q * 8) = *(const v8bf*)(g + (long)r * ld + q * 8);
  }
}

// TDM (TENSORcnt-tracked) stage of rows x 32 bf16 2D tile, row stride ld (elems).
// D# per CDNA5 ISA: data_size=2B, tile_dim0=32, tile_dim1=rows,
// LDS padding 4 DW every 16 DW -> LDS row stride 40 elems == LS.
__device__ inline void tdm_stage(bf16_t* lds, const bf16_t* g, long ld_elems, int rows) {
#ifdef HAVE_TDM
  unsigned long long ga = (unsigned long long)(size_t)g;
  unsigned int la = (unsigned int)(size_t)lds;
  u32x4 g0;
  g0[0] = 1u;                                                  // count=1, user mode
  g0[1] = la;                                                  // lds_addr
  g0[2] = (unsigned int)ga;                                    // global_addr[31:0]
  g0[3] = (unsigned int)((ga >> 32) & 0x1FFFFFFu) | (2u << 30); // addr[56:32], type=2
  const unsigned int dim0 = 1u << 30, dim1 = 1u << 30;         // huge tensor: no OOB clip
  unsigned long long st = (unsigned long long)ld_elems;
  i32x8 g1;
  g1[0] = (int)((1u << 16) | (1u << 20) | (3u << 22) | (3u << 25)); // data_size=2B,pad_en,int=16DW,amt=4DW
  g1[1] = (int)((dim0 & 0xFFFFu) << 16);
  g1[2] = (int)((dim0 >> 16) | ((dim1 & 0xFFFFu) << 16));
  g1[3] = (int)((dim1 >> 16) | (32u << 16));                   // tile_dim0 = 32
  g1[4] = (int)(rows & 0xFFFF);                                // tile_dim1 = rows, tile_dim2 = 0
  g1[5] = (int)(unsigned int)st;                               // dim0 stride lo
  g1[6] = (int)((st >> 32) & 0xFFFFu);                         // stride hi, dim1 stride = 0
  g1[7] = 0;
  i32x4 z4 = {0, 0, 0, 0};
  i32x8 z8 = {0, 0, 0, 0, 0, 0, 0, 0};
  __builtin_amdgcn_tensor_load_to_lds(g0, g1, z4, z4, z8, 0);
#else
  stage_sync(lds, g, ld_elems, rows, threadIdx.x & 255);
#endif
}

__device__ inline void wait_tensor() {
#ifdef HAVE_TDM
  __builtin_amdgcn_s_wait_tensorcnt(0);
#endif
}

// ---------------------------------------------------------------------------
// Pre-pass conversion kernels
// ---------------------------------------------------------------------------
__global__ __launch_bounds__(256) void k_cvt(const float* __restrict__ in,
                                             bf16_t* __restrict__ out, long n) {
  long stride = (long)gridDim.x * 256;
  for (long i = (long)blockIdx.x * 256 + threadIdx.x; i < n; i += stride)
    out[i] = (bf16_t)in[i];
}

// Tiled transpose-convert: in[R][C] fp32 -> out[C][R] bf16.  Grid (C/32, R/32).
__global__ __launch_bounds__(256) void k_tr(const float* __restrict__ in,
                                            bf16_t* __restrict__ out, int R, int C) {
  __shared__ float tile[32][33];
  int rb = blockIdx.y * 32, cb = blockIdx.x * 32;
  int tx = threadIdx.x & 31, ty = threadIdx.x >> 5;  // 32 x 8
#pragma unroll
  for (int i = 0; i < 32; i += 8)
    tile[ty + i][tx] = in[(long)(rb + ty + i) * C + cb + tx];
  __syncthreads();
#pragma unroll
  for (int i = 0; i < 32; i += 8)
    out[(long)(cb + ty + i) * R + rb + tx] = (bf16_t)tile[tx][ty + i];
}

// A[NA][Din][16] fp32 -> out[128][Din] bf16 : out[(a*16+r)*Din + d] = A[a][d][r]
__global__ __launch_bounds__(256) void k_acat(const float* __restrict__ A,
                                              bf16_t* __restrict__ out, int Din) {
  long n = (long)NR * Din;
  long stride = (long)gridDim.x * 256;
  for (long i = (long)blockIdx.x * 256 + threadIdx.x; i < n; i += stride) {
    long row = i / Din, d = i % Din;
    long a = row >> 4, r = row & 15;
    out[i] = (bf16_t)A[(a * (long)Din + d) * 16 + r];
  }
}

// ---------------------------------------------------------------------------
// Rank projection: hexp[T,128] = mask(Abf[T,K] @ AcatT[128,K]^T).
// Block 128x128, 8 waves (2m x 4n), 4x2 acc tiles, double-buffered async/TDM.
// ---------------------------------------------------------------------------
__global__ __launch_bounds__(256)
void k_rank(const bf16_t* __restrict__ Abf, long lda,
            const bf16_t* __restrict__ AcatT,
            const int* __restrict__ idx,
            bf16_t* __restrict__ hexp, int K) {
  __shared__ bf16_t Alds[2][128 * LS];
  __shared__ bf16_t Blds[2][128 * LS];
  const int tid = threadIdx.x, lane = tid & 31, wave = tid >> 5;
  const int m0 = blockIdx.x * 128;
  const int wm = (wave >> 2) * 64;
  const int wn = (wave & 3) * 32;
  v8f acc[4][2] = {};
  const int P = K / 32;

  auto stage = [&](int p, int buf) {
    stage_async(&Alds[buf][0], Abf + (long)m0 * lda + p * 32, lda, 128, tid);
    if (wave == 0) tdm_stage(&Blds[buf][0], AcatT + p * 32, K, 128);
  };

  int buf = 0;
  stage(0, 0);
  wait_async();
  if (wave == 0) wait_tensor();
  __syncthreads();
  for (int p = 0; p < P; ++p) {
    if (p + 1 < P) stage(p + 1, buf ^ 1);
    v16bf b[2];
#pragma unroll
    for (int ni = 0; ni < 2; ++ni) b[ni] = load_b_frag(&Blds[buf][0], wn + ni * 16, lane);
#pragma unroll
    for (int mi = 0; mi < 4; ++mi) {
      v16bf a = load_a_frag(&Alds[buf][0], wm + mi * 16, lane);
#pragma unroll
      for (int ni = 0; ni < 2; ++ni) acc[mi][ni] = wmma_bf16(a, b[ni], acc[mi][ni]);
    }
    if (p + 1 < P) {
      wait_async();
      if (wave == 0) wait_tensor();
      __syncthreads();
    }
    buf ^= 1;
  }

#pragma unroll
  for (int mi = 0; mi < 4; ++mi)
#pragma unroll
    for (int ni = 0; ni < 2; ++ni) {
      int rbase = m0 + wm + mi * 16 + ((lane >> 4) << 3);
      int col   = wn + ni * 16 + (lane & 15);
#pragma unroll
      for (int r = 0; r < 8; ++r) {
        int t = rbase + r;
        bool keep = ((col >> 4) == idx[t]);
        hexp[(long)t * NR + col] = keep ? (bf16_t)acc[mi][ni][r] : (bf16_t)0.0f;
      }
    }
}

// ---------------------------------------------------------------------------
// Fused gate/up GEMM + LoRA K-extension + SiLU*up epilogue -> hidden bf16.
// Block 128 rows x 64 hidden cols; 8 waves (4m x 2n); 2x2 tiles each for gate/up.
// ---------------------------------------------------------------------------
__global__ __launch_bounds__(256)
void k_gateup(const bf16_t* __restrict__ xb, const bf16_t* __restrict__ WguT,
              const bf16_t* __restrict__ hgu, const bf16_t* __restrict__ BguT,
              bf16_t* __restrict__ hidden) {
  __shared__ bf16_t Alds[2][128 * LS];
  __shared__ bf16_t Bg[2][64 * LS];
  __shared__ bf16_t Bu[2][64 * LS];
  const int tid = threadIdx.x, lane = tid & 31, wave = tid >> 5;
  const int m0 = blockIdx.y * 128;
  const int n0 = blockIdx.x * 64;
  const int wm = (wave >> 1) * 32;
  const int wn = (wave & 1) * 32;
  v8f accg[2][2] = {}, accu[2][2] = {};
  const int P1 = DIM_D / 32, P = P1 + NR / 32;

  auto stage = [&](int p, int buf) {
    const bf16_t* Ap; const bf16_t* Bp; long lda, ldb; int k0;
    if (p < P1) { k0 = p * 32;        Ap = xb;  lda = DIM_D; Bp = WguT; ldb = DIM_D; }
    else        { k0 = (p - P1) * 32; Ap = hgu; lda = NR;    Bp = BguT; ldb = NR; }
    stage_async(&Alds[buf][0], Ap + (long)m0 * lda + k0, lda, 128, tid);
    if (wave == 0) {
      tdm_stage(&Bg[buf][0], Bp + (long)n0 * ldb + k0, ldb, 64);
      tdm_stage(&Bu[buf][0], Bp + (long)(DIM_I + n0) * ldb + k0, ldb, 64);
    }
  };

  int buf = 0;
  stage(0, 0);
  wait_async();
  if (wave == 0) wait_tensor();
  __syncthreads();
  for (int p = 0; p < P; ++p) {
    if (p + 1 < P) stage(p + 1, buf ^ 1);
    v16bf bg[2], bu[2];
#pragma unroll
    for (int ni = 0; ni < 2; ++ni) {
      bg[ni] = load_b_frag(&Bg[buf][0], wn + ni * 16, lane);
      bu[ni] = load_b_frag(&Bu[buf][0], wn + ni * 16, lane);
    }
#pragma unroll
    for (int mi = 0; mi < 2; ++mi) {
      v16bf a = load_a_frag(&Alds[buf][0], wm + mi * 16, lane);
#pragma unroll
      for (int ni = 0; ni < 2; ++ni) {
        accg[mi][ni] = wmma_bf16(a, bg[ni], accg[mi][ni]);
        accu[mi][ni] = wmma_bf16(a, bu[ni], accu[mi][ni]);
      }
    }
    if (p + 1 < P) {
      wait_async();
      if (wave == 0) wait_tensor();
      __syncthreads();
    }
    buf ^= 1;
  }

#pragma unroll
  for (int mi = 0; mi < 2; ++mi)
#pragma unroll
    for (int ni = 0; ni < 2; ++ni) {
      int rbase = m0 + wm + mi * 16 + ((lane >> 4) << 3);
      int col   = n0 + wn + ni * 16 + (lane & 15);
#pragma unroll
      for (int r = 0; r < 8; ++r) {
        float g = accg[mi][ni][r];
        float u = accu[mi][ni][r];
        float s = g / (1.0f + __expf(-g));   // SiLU
        hidden[(long)(rbase + r) * DIM_I + col] = (bf16_t)(s * u);
      }
    }
}

// ---------------------------------------------------------------------------
// Down projection: out[T,D] f32 = hidden @ W_down + hdn @ B_down.
// Block 128x128; 8 waves (2m x 4n); 4x2 acc tiles.
// ---------------------------------------------------------------------------
__global__ __launch_bounds__(256)
void k_down(const bf16_t* __restrict__ hid, const bf16_t* __restrict__ WdT,
            const bf16_t* __restrict__ hdn, const bf16_t* __restrict__ BdT,
            float* __restrict__ out) {
  __shared__ bf16_t Alds[2][128 * LS];
  __shared__ bf16_t Blds[2][128 * LS];
  const int tid = threadIdx.x, lane = tid & 31, wave = tid >> 5;
  const int m0 = blockIdx.y * 128;
  const int n0 = blockIdx.x * 128;
  const int wm = (wave >> 2) * 64;
  const int wn = (wave & 3) * 32;
  v8f acc[4][2] = {};
  const int P1 = DIM_I / 32, P = P1 + NR / 32;

  auto stage = [&](int p, int buf) {
    const bf16_t* Ap; const bf16_t* Bp; long lda, ldb; int k0;
    if (p < P1) { k0 = p * 32;        Ap = hid; lda = DIM_I; Bp = WdT; ldb = DIM_I; }
    else        { k0 = (p - P1) * 32; Ap = hdn; lda = NR;    Bp = BdT; ldb = NR; }
    stage_async(&Alds[buf][0], Ap + (long)m0 * lda + k0, lda, 128, tid);
    if (wave == 0) tdm_stage(&Blds[buf][0], Bp + (long)n0 * ldb + k0, ldb, 128);
  };

  int buf = 0;
  stage(0, 0);
  wait_async();
  if (wave == 0) wait_tensor();
  __syncthreads();
  for (int p = 0; p < P; ++p) {
    if (p + 1 < P) stage(p + 1, buf ^ 1);
    v16bf b[2];
#pragma unroll
    for (int ni = 0; ni < 2; ++ni) b[ni] = load_b_frag(&Blds[buf][0], wn + ni * 16, lane);
#pragma unroll
    for (int mi = 0; mi < 4; ++mi) {
      v16bf a = load_a_frag(&Alds[buf][0], wm + mi * 16, lane);
#pragma unroll
      for (int ni = 0; ni < 2; ++ni) acc[mi][ni] = wmma_bf16(a, b[ni], acc[mi][ni]);
    }
    if (p + 1 < P) {
      wait_async();
      if (wave == 0) wait_tensor();
      __syncthreads();
    }
    buf ^= 1;
  }

#pragma unroll
  for (int mi = 0; mi < 4; ++mi)
#pragma unroll
    for (int ni = 0; ni < 2; ++ni) {
      int rbase = m0 + wm + mi * 16 + ((lane >> 4) << 3);
      int col   = n0 + wn + ni * 16 + (lane & 15);
#pragma unroll
      for (int r = 0; r < 8; ++r)
        out[(long)(rbase + r) * DIM_D + col] = acc[mi][ni][r];
    }
}

// ---------------------------------------------------------------------------
extern "C" void kernel_launch(void* const* d_in, const int* in_sizes, int n_in,
                              void* d_out, int out_size, void* d_ws, size_t ws_size,
                              hipStream_t stream) {
  const float* x   = (const float*)d_in[0];
  const int*   idx = (const int*)d_in[1];
  const float* Wgu = (const float*)d_in[2];
  const float* Agu = (const float*)d_in[3];
  const float* Bgu = (const float*)d_in[4];
  const float* Wdn = (const float*)d_in[5];
  const float* Adn = (const float*)d_in[6];
  const float* Bdn = (const float*)d_in[7];
  float* out = (float*)d_out;

  char* ws = (char*)d_ws;
  size_t off = 0;
  auto alloc = [&](size_t bytes) -> void* {
    void* p = ws + off;
    off = (off + bytes + 255) & ~(size_t)255;
    return p;
  };

  bf16_t* xb   = (bf16_t*)alloc((size_t)T_TOK * DIM_D * 2);
  bf16_t* WguT = (bf16_t*)alloc((size_t)TWO_I * DIM_D * 2);
  bf16_t* WdnT = (bf16_t*)alloc((size_t)DIM_D * DIM_I * 2);
  bf16_t* BguT = (bf16_t*)alloc((size_t)TWO_I * NR * 2);
  bf16_t* BdnT = (bf16_t*)alloc((size_t)DIM_D * NR * 2);
  bf16_t* AguC = (bf16_t*)alloc((size_t)NR * DIM_D * 2);
  bf16_t* AdnC = (bf16_t*)alloc((size_t)NR * DIM_I * 2);
  bf16_t* hgu  = (bf16_t*)alloc((size_t)T_TOK * NR * 2);
  bf16_t* hdn  = (bf16_t*)alloc((size_t)T_TOK * NR * 2);
  bf16_t* hid  = (bf16_t*)alloc((size_t)T_TOK * DIM_I * 2);

  // Pre-pass: fp32 -> bf16 (+ tiled transpose of weights into N-major)
  k_cvt <<<4096, 256, 0, stream>>>(x, xb, (long)T_TOK * DIM_D);
  k_tr  <<<dim3(TWO_I / 32, DIM_D / 32), 256, 0, stream>>>(Wgu, WguT, DIM_D, TWO_I);
  k_tr  <<<dim3(DIM_D / 32, DIM_I / 32), 256, 0, stream>>>(Wdn, WdnT, DIM_I, DIM_D);
  k_tr  <<<dim3(TWO_I / 32, NR / 32),    256, 0, stream>>>(Bgu, BguT, NR, TWO_I);
  k_tr  <<<dim3(DIM_D / 32, NR / 32),    256, 0, stream>>>(Bdn, BdnT, NR, DIM_D);
  k_acat<<<512,  256, 0, stream>>>(Agu, AguC, DIM_D);
  k_acat<<<1024, 256, 0, stream>>>(Adn, AdnC, DIM_I);

  // LoRA rank projection for gate/up: hgu = mask(x @ A_cat)
  k_rank<<<dim3(T_TOK / 128), 256, 0, stream>>>(xb, DIM_D, AguC, idx, hgu, DIM_D);

  // Fused gate/up GEMM (base + LoRA K-ext) -> SiLU*up -> hidden (bf16)
  k_gateup<<<dim3(DIM_I / 64, T_TOK / 128), 256, 0, stream>>>(xb, WguT, hgu, BguT, hid);

  // LoRA rank projection for down: hdn = mask(hidden @ A_cat_down)
  k_rank<<<dim3(T_TOK / 128), 256, 0, stream>>>(hid, DIM_I, AdnC, idx, hdn, DIM_I);

  // Down projection (base + LoRA K-ext) -> out (f32)
  k_down<<<dim3(DIM_D / 128, T_TOK / 128), 256, 0, stream>>>(hid, WdnT, hdn, BdnT, out);
}
